// HoughVoteLayer_23871428231214
// MI455X (gfx1250) — matrix operand order repaired and checked by reading, not compiled, not run on hardware
//
#include <hip/hip_runtime.h>
#include <hip/hip_bf16.h>

typedef __attribute__((ext_vector_type(16))) _Float16 v16h;
typedef __attribute__((ext_vector_type(8)))  float    v8f;

#define HH 240
#define WW 320
#define PP (HH*WW)          // 76800
#define NC 22
#define QN 300              // 15*20 query points
#define QW 20               // queries per row
#define ACCROWS 320
#define ACCCOLS 32
#define EPSF 1e-6f
#define INLIER_T 0.9f
#define PSEGS 75            // P-dim split: 2400 chunks / 75 = 32 chunks each

// ---- workspace layout (floats) ----
#define OFF_UX   0
#define OFF_UY   (PP)
#define OFF_DZ   (2*PP)
#define OFF_LAB  (3*PP)
#define OFF_ACC  (4*PP)
#define OFF_CLS  (4*PP + ACCROWS*ACCCOLS)
#define OFF_BV   (OFF_CLS + NC)
#define OFF_BQ   (OFF_BV + NC)
#define OFF_CNT  (OFF_BQ + NC)
#define OFF_ZS   (OFF_CNT + NC)
#define ZERO_N   (ACCROWS*ACCCOLS + 5*NC + 32)

__global__ void hv_zero(float* ws) {
    int i = blockIdx.x * blockDim.x + threadIdx.x;
    if (i < ZERO_N) ws[OFF_ACC + i] = 0.0f;
}

__global__ void hv_prep(const int* __restrict__ label,
                        const float* __restrict__ vp,
                        float* __restrict__ ws) {
    int p = blockIdx.x * blockDim.x + threadIdx.x;
    if (p >= PP) return;
    int lab = label[p];
    float d0 = vp[(3*lab + 0) * PP + p];
    float d1 = vp[(3*lab + 1) * PP + p];
    float d2 = vp[(3*lab + 2) * PP + p];
    float dn = sqrtf(d0*d0 + d1*d1) + EPSF;
    bool fg = lab > 0;
    ws[OFF_UX + p] = fg ? d0 / dn : 0.0f;   // fg folded: lab==0 -> never inlier
    ws[OFF_UY + p] = fg ? d1 / dn : 0.0f;
    ws[OFF_DZ + p] = d2;
    ((int*)ws)[OFF_LAB + p] = lab;
    atomicAdd(&ws[OFF_CLS + lab], 1.0f);
}

// WMMA voting: acc[q,c] += sum_p inlier[q,p] * onehot[p,c]
// grid.x = 5 query-groups (4 waves each -> 20 tiles of 16 q)
// grid.y = PSEGS P-segments (32 chunks of 32 pixels each)
__global__ void __launch_bounds__(128)
hv_vote(float* __restrict__ ws) {
    __shared__ float sux[32], suy[32], spx[32], spy[32];
    __shared__ int   slab[32];

    const int tid  = threadIdx.x;
    const int lane = tid & 31;
    const int wid  = tid >> 5;          // 0..3 (wave32)
    const int hi   = lane >> 4;         // lane group
    const int nl   = lane & 15;

    const int tile = blockIdx.x * 4 + wid;     // 0..19
    const int q    = tile * 16 + nl;           // A-row query for this lane
    const float qxL = (float)((q % QW) * 16);
    const float qyL = (float)((q / QW) * 16);

    const float* g_ux = ws + OFF_UX;
    const float* g_uy = ws + OFF_UY;
    const int*   g_lab = ((const int*)ws) + OFF_LAB;

    v8f acc0 = {}; v8f acc1 = {};

    const int kc0 = blockIdx.y * (2400 / PSEGS);
    const int kc1 = kc0 + (2400 / PSEGS);
    for (int kc = kc0; kc < kc1; ++kc) {
        const int pbase = kc * 32;
        __syncthreads();
        if (tid < 32) {
            int pix = pbase + tid;
            sux[tid] = g_ux[pix];
            suy[tid] = g_uy[pix];
            slab[tid] = g_lab[pix];
            spx[tid] = (float)(pix % WW);
            spy[tid] = (float)(pix / WW);
        }
        __syncthreads();

        // Build B first (32x16 f16 one-hot): lane group hi covers K = hi*16 + e.
        // Keeping both builds before the WMMAs maximizes the distance between a
        // WMMA read of A/B and their rewrite next iteration (hides the 4-coexec
        // WAR hazard behind the barrier + staging instead of v_nops).
        v16h b0, b1;
#pragma unroll
        for (int e = 0; e < 16; ++e) {
            int lb = slab[hi * 16 + e];
            b0[e] = (_Float16)((lb == nl)      ? 1.0f : 0.0f);
            b1[e] = (_Float16)((lb == nl + 16) ? 1.0f : 0.0f);
        }
        // Build A (16x32 f16): lane group hi covers K = hi*16 + e
        v16h a;
#pragma unroll
        for (int e = 0; e < 16; ++e) {
            int K = hi * 16 + e;
            float ux = sux[K], uy = suy[K];
            float cx = qxL - spx[K];
            float cy = qyL - spy[K];
            float dot = cx * ux + cy * uy;
            float cn  = sqrtf(cx * cx + cy * cy) + EPSF;
            a[e] = (_Float16)((dot > INLIER_T * cn) ? 1.0f : 0.0f);
        }
        acc0 = __builtin_amdgcn_wmma_f32_16x16x32_f16(false, a, false, b0,
                                                      (short)0, acc0, false, false);
        acc1 = __builtin_amdgcn_wmma_f32_16x16x32_f16(false, a, false, b1,
                                                      (short)0, acc1, false, false);
    }

    // D layout: vgpr j, lane -> row M = j + hi*8, col N = lane&15
    float* acc = ws + OFF_ACC;
#pragma unroll
    for (int j = 0; j < 8; ++j) {
        int row = tile * 16 + j + hi * 8;
        atomicAdd(&acc[row * ACCCOLS + nl],      acc0[j]);
        atomicAdd(&acc[row * ACCCOLS + 16 + nl], acc1[j]);
    }
}

// per-class argmax over q<300 (first-occurrence tie-break)
__global__ void __launch_bounds__(64)
hv_argbest(float* __restrict__ ws) {
    __shared__ float sval[64];
    __shared__ int   sidx[64];
    const int c = blockIdx.x;
    const int t = threadIdx.x;
    const float* acc = ws + OFF_ACC;
    float bv = -1.0f; int bq = 0;
    for (int q = t; q < QN; q += 64) {
        float v = acc[q * ACCCOLS + c];
        if (v > bv || (v == bv && q < bq)) { bv = v; bq = q; }
    }
    sval[t] = bv; sidx[t] = bq;
    __syncthreads();
    for (int s = 32; s > 0; s >>= 1) {
        if (t < s) {
            float ov = sval[t + s]; int oi = sidx[t + s];
            if (ov > sval[t] || (ov == sval[t] && oi < sidx[t])) {
                sval[t] = ov; sidx[t] = oi;
            }
        }
        __syncthreads();
    }
    if (t == 0) {
        ws[OFF_BV + c] = sval[0];
        ((int*)ws)[OFF_BQ + c] = sidx[0];
    }
}

// second sweep: cnt[c], zsum[c] for q = best_q[c]
__global__ void __launch_bounds__(256)
hv_zpass(float* __restrict__ ws) {
    __shared__ float scnt[256], szs[256];
    const int c = blockIdx.x;
    const int t = threadIdx.x;
    const int bq = ((const int*)ws)[OFF_BQ + c];
    const float qx = (float)((bq % QW) * 16);
    const float qy = (float)((bq / QW) * 16);
    const float* g_ux = ws + OFF_UX;
    const float* g_uy = ws + OFF_UY;
    const float* g_dz = ws + OFF_DZ;
    const int*   g_lab = ((const int*)ws) + OFF_LAB;
    float cnt = 0.0f, zs = 0.0f;
    for (int p = t; p < PP; p += 256) {
        if (g_lab[p] != c) continue;
        float cx = qx - (float)(p % WW);
        float cy = qy - (float)(p / WW);
        float dot = cx * g_ux[p] + cy * g_uy[p];
        float cn  = sqrtf(cx * cx + cy * cy) + EPSF;
        if (dot > INLIER_T * cn) { cnt += 1.0f; zs += g_dz[p]; }
    }
    scnt[t] = cnt; szs[t] = zs;
    __syncthreads();
    for (int s = 128; s > 0; s >>= 1) {
        if (t < s) { scnt[t] += scnt[t + s]; szs[t] += szs[t + s]; }
        __syncthreads();
    }
    if (t == 0) { ws[OFF_CNT + c] = scnt[0]; ws[OFF_ZS + c] = szs[0]; }
}

__global__ void hv_finalize(const float* __restrict__ extents,
                            const float* __restrict__ poses,
                            const float* __restrict__ meta,
                            const float* __restrict__ ws,
                            float* __restrict__ out) {
    const int c = threadIdx.x;
    if (c >= NC) return;
    const float fx = meta[0] + EPSF, ppx = meta[2];
    const float fy = meta[4] + EPSF, ppy = meta[5];
    const float e0 = extents[c*3+0], e1 = extents[c*3+1], e2 = extents[c*3+2];
    const float half = 0.5f * sqrtf(e0*e0 + e1*e1 + e2*e2);
    const float bv = ws[OFF_BV + c];
    const float cc = ws[OFF_CLS + c];
    const int   bq = ((const int*)ws)[OFF_BQ + c];
    const float bx = (float)((bq % QW) * 16);
    const float by = (float)((bq / QW) * 16);
    const float z  = ws[OFF_ZS + c] / (ws[OFF_CNT + c] + EPSF);
    const float zsafe = (fabsf(z) > EPSF) ? z : EPSF;
    const float r = fx * half / zsafe;
    const bool valid = (bv >= 50.0f) && (cc >= 500.0f) && (bv / (cc + EPSF) >= 0.02f);
    const float score = valid ? bv : 0.0f;
    const float tx = (bx - ppx) * z / fx;
    const float ty = (by - ppy) * z / fy;
    float* o = out + c * 14;
    o[0] = 0.0f;       o[1] = (float)c;
    o[2] = bx - r;     o[3] = by - r;
    o[4] = bx + r;     o[5] = by + r;
    o[6] = score;
    o[7]  = poses[c*13 + 6];
    o[8]  = poses[c*13 + 7];
    o[9]  = poses[c*13 + 8];
    o[10] = poses[c*13 + 9];
    o[11] = tx; o[12] = ty; o[13] = z;
}

extern "C" void kernel_launch(void* const* d_in, const int* in_sizes, int n_in,
                              void* d_out, int out_size, void* d_ws, size_t ws_size,
                              hipStream_t stream) {
    const int*   label   = (const int*)d_in[0];
    const float* vp      = (const float*)d_in[1];
    const float* extents = (const float*)d_in[2];
    const float* poses   = (const float*)d_in[3];
    const float* meta    = (const float*)d_in[4];
    float* out = (float*)d_out;
    float* ws  = (float*)d_ws;

    hv_zero<<<(ZERO_N + 255) / 256, 256, 0, stream>>>(ws);
    hv_prep<<<(PP + 255) / 256, 256, 0, stream>>>(label, vp, ws);
    hv_vote<<<dim3(5, PSEGS), 128, 0, stream>>>(ws);
    hv_argbest<<<NC, 64, 0, stream>>>(ws);
    hv_zpass<<<NC, 256, 0, stream>>>(ws);
    hv_finalize<<<1, 32, 0, stream>>>(extents, poses, meta, ws, out);
}
